// SuperPoint_3917010174433
// MI455X (gfx1250) — matrix operand order repaired and checked by reading, not compiled, not run on hardware
//
#include <hip/hip_runtime.h>
#include <math.h>

typedef _Float16 half16 __attribute__((ext_vector_type(16)));
typedef _Float16 half8  __attribute__((ext_vector_type(8)));
typedef float    float8 __attribute__((ext_vector_type(8)));

union AF { half16 v; _Float16 h[16]; };
union CF { float8 v; float f[8]; };
union H8 { half8 v; _Float16 h[8]; };

#define SCORE_THR 0.005f
#define CDIV(a,b) (((a)+(b)-1)/(b))

// ---------------------------------------------------------------- utilities
__global__ void k_f32_to_f16(const float* __restrict__ in, _Float16* __restrict__ out, int n){
  int i = blockIdx.x*blockDim.x + threadIdx.x;
  if (i < n) out[i] = (_Float16)in[i];
}

// Repack OIHW f32 weights into per-lane WMMA A-fragment layout (f16).
// K ordering: k = s*Cin + ci  (s = ky*3+kx spatial tap, channel-minor) so that
// B fragments are contiguous in NHWC activations.
// A 16x32 f16 packing used here (same k order as B): M = lane%16;  K = e + 16*(lane>>4).
__global__ void k_pack_w(const float* __restrict__ w, _Float16* __restrict__ dst,
                         int Cout, int Cin, int KK, int ksteps, int total){
  int tid = blockIdx.x*blockDim.x + threadIdx.x;
  if (tid >= total) return;
  int K  = Cin*KK;
  int e  = tid & 15;
  int L  = (tid >> 4) & 31;
  int fs = tid >> 9;
  int mt = fs / ksteps;
  int ks = fs - mt*ksteps;
  int j  = e >> 1;
  int k  = ks*32 + e + ((L>>4)<<4);
  int m  = mt*16 + (L & 15);
  float v = 0.f;
  if (m < Cout && k < K){
    int s  = k / Cin;           // spatial tap
    int ci = k - s*Cin;         // input channel
    v = w[(size_t)m*K + (size_t)ci*KK + s];   // OIHW flat: m*Cin*KK + ci*KK + s
  }
  dst[tid] = (_Float16)v;
}

// Implicit-GEMM conv (3x3 pad1 or 1x1) via v_wmma_f32_16x16x32_f16.
// Activations NHWC f16. Each wave: one 16-row M tile x four 16-pixel N tiles
// (A fragment reused across all four WMMAs per K step).
// grid.x = H*W/64, grid.y = ceil(Cout/16), block = 32 (one wave).
__global__ __launch_bounds__(32)
void k_conv_wmma(const _Float16* __restrict__ in, const _Float16* __restrict__ packW,
                 const float* __restrict__ bias, float* __restrict__ out32,
                 _Float16* __restrict__ out16,
                 int Cin, int Cout, int H, int W, int KK, int padc, int ksteps,
                 int relu, int fast){
  int L   = threadIdx.x;
  int mt  = blockIdx.y;
  int lp  = L & 15;
  int pix[4];
  pix[0] = blockIdx.x*64 + lp;
  pix[1] = pix[0] + 16;
  pix[2] = pix[0] + 32;
  pix[3] = pix[0] + 48;
  int px[4], py[4];
#pragma unroll
  for (int t = 0; t < 4; ++t){ px[t] = pix[t] % W; py[t] = pix[t] / W; }
  int K   = Cin * KK;
  int hiK = (L >> 4) << 4;              // B frag: upper-half lanes hold K+16..K+31
  int mr  = (L >> 4) << 3;              // C/D frag: upper-half lanes hold M+8..M+15

  AF zf;
#pragma unroll
  for (int e = 0; e < 16; ++e) zf.h[e] = (_Float16)0.f;

  // branchless bias preload (clamped index, select-zero)
  CF acc[4];
#pragma unroll
  for (int r = 0; r < 8; ++r){
    int m = mt*16 + r + mr;
    int mc = m < (Cout-1) ? m : (Cout-1);
    float bv = bias[mc];
    if (m >= Cout) bv = 0.f;
    acc[0].f[r] = bv; acc[1].f[r] = bv; acc[2].f[r] = bv; acc[3].f[r] = bv;
  }

  const half16* ap = reinterpret_cast<const half16*>(packW) + (size_t)mt*ksteps*32 + L;
#pragma unroll 2
  for (int ks = 0; ks < ksteps; ++ks){
    half16 a = ap[(size_t)ks * 32];
    AF b[4];
    if (fast){
      // whole 32-wide K step lies inside one spatial tap (Cin % 32 == 0)
      int kbase = ks*32;
      int s   = kbase / Cin;
      int ci0 = kbase - s*Cin + hiK;
      int ky  = s / 3, kx = s - ky*3;           // s==0 for 1x1
#pragma unroll
      for (int t = 0; t < 4; ++t){
        int yy = py[t] + ky - padc, xx = px[t] + kx - padc;
        if (yy >= 0 && yy < H && xx >= 0 && xx < W)
          b[t].v = *reinterpret_cast<const half16*>(in + ((size_t)yy*W + xx)*Cin + ci0);
        else
          b[t] = zf;
      }
    } else {
      // generic gather (layer 1a only: Cin==1, single K step)
#pragma unroll
      for (int t = 0; t < 4; ++t) b[t] = zf;
#pragma unroll
      for (int e = 0; e < 16; ++e){
        int k = ks*32 + hiK + e;
        if (k < K){
          int s  = k / Cin;
          int ci = k - s*Cin;
          int ky = s / 3, kx = s - ky*3;
#pragma unroll
          for (int t = 0; t < 4; ++t){
            int yy = py[t] + ky - padc, xx = px[t] + kx - padc;
            if (yy >= 0 && yy < H && xx >= 0 && xx < W)
              b[t].h[e] = in[((size_t)yy*W + xx)*Cin + ci];
          }
        }
      }
    }
    acc[0].v = __builtin_amdgcn_wmma_f32_16x16x32_f16(false, a, false, b[0].v,
                                                      (short)0, acc[0].v, false, false);
    acc[1].v = __builtin_amdgcn_wmma_f32_16x16x32_f16(false, a, false, b[1].v,
                                                      (short)0, acc[1].v, false, false);
    acc[2].v = __builtin_amdgcn_wmma_f32_16x16x32_f16(false, a, false, b[2].v,
                                                      (short)0, acc[2].v, false, false);
    acc[3].v = __builtin_amdgcn_wmma_f32_16x16x32_f16(false, a, false, b[3].v,
                                                      (short)0, acc[3].v, false, false);
  }

  if (out16){
    // f16 path: Cout is 64/128/256 -> lane's 8 rows are 8 consecutive channels,
    // 16-byte aligned -> one half8 store per pixel tile.
    int m0 = mt*16 + mr;
#pragma unroll
    for (int t = 0; t < 4; ++t){
      H8 o;
#pragma unroll
      for (int r = 0; r < 8; ++r){
        float v = acc[t].f[r];
        if (relu) v = fmaxf(v, 0.f);
        o.h[r] = (_Float16)v;
      }
      *reinterpret_cast<half8*>(out16 + (size_t)pix[t]*Cout + m0) = o.v;
    }
  } else {
    // f32 heads (NCHW), Cout may be ragged (65)
#pragma unroll
    for (int r = 0; r < 8; ++r){
      int m = mt*16 + r + mr;
      if (m < Cout){
#pragma unroll
        for (int t = 0; t < 4; ++t){
          float v = acc[t].f[r];
          if (relu) v = fmaxf(v, 0.f);
          out32[(size_t)m*H*W + pix[t]] = v;
        }
      }
    }
  }
}

// 2x2 max pool, NHWC f16
__global__ void k_pool2(const _Float16* __restrict__ in, _Float16* __restrict__ out,
                        int C, int Ho, int Wo){
  int i = blockIdx.x*blockDim.x + threadIdx.x;
  int n = C*Ho*Wo;
  if (i >= n) return;
  int c = i % C;
  int p = i / C;
  int y = p / Wo, x = p - y*Wo;
  int Wi = Wo*2;
  const _Float16* b = in + (((size_t)(2*y)*Wi) + 2*x)*C + c;
  size_t rs = (size_t)Wi*C;
  float m = fmaxf(fmaxf((float)b[0], (float)b[C]),
                  fmaxf((float)b[rs], (float)b[rs + C]));
  out[i] = (_Float16)m;
}

// softmax over 65 channels (NCHW f32), drop last, depth-to-space into 480x640 heatmap
__global__ void k_softmax65(const float* __restrict__ s0, float* __restrict__ heat){
  int p = blockIdx.x*blockDim.x + threadIdx.x;
  if (p >= 4800) return;
  int y = p / 80, x = p - y*80;
  float mx = -1e30f;
  for (int c = 0; c < 65; ++c) mx = fmaxf(mx, s0[c*4800 + p]);
  float sum = 0.f;
  for (int c = 0; c < 65; ++c) sum += expf(s0[c*4800 + p] - mx);
  float inv = 1.f / sum;
  for (int c = 0; c < 64; ++c){
    float v = expf(s0[c*4800 + p] - mx) * inv;
    heat[((y<<3) + (c>>3))*640 + (x<<3) + (c&7)] = v;
  }
}

__global__ void k_mp_row(const float* __restrict__ in, float* __restrict__ out){
  int i = blockIdx.x*blockDim.x + threadIdx.x;
  if (i >= 480*640) return;
  int y = i / 640, x = i - y*640;
  int lo = x-4; if (lo < 0) lo = 0;
  int hi = x+4; if (hi > 639) hi = 639;
  float m = -1e30f;
  for (int xx = lo; xx <= hi; ++xx) m = fmaxf(m, in[y*640 + xx]);
  out[i] = m;
}

__global__ void k_mp_col(const float* __restrict__ in, float* __restrict__ out){
  int i = blockIdx.x*blockDim.x + threadIdx.x;
  if (i >= 480*640) return;
  int y = i / 640, x = i - y*640;
  int lo = y-4; if (lo < 0) lo = 0;
  int hi = y+4; if (hi > 479) hi = 479;
  float m = -1e30f;
  for (int yy = lo; yy <= hi; ++yy) m = fmaxf(m, in[yy*640 + x]);
  out[i] = m;
}

__global__ void k_eq_mask(const float* __restrict__ a, const float* __restrict__ b,
                          float* __restrict__ mask, int n){
  int i = blockIdx.x*blockDim.x + threadIdx.x;
  if (i < n) mask[i] = (a[i] == b[i]) ? 1.f : 0.f;
}

__global__ void k_supp_ss(const float* __restrict__ suppmax, const float* __restrict__ heat,
                          float* __restrict__ ss, int n){
  int i = blockIdx.x*blockDim.x + threadIdx.x;
  if (i < n) ss[i] = (suppmax[i] > 0.f) ? 0.f : heat[i];
}

__global__ void k_mask_update(float* __restrict__ mask, const float* __restrict__ ss,
                              const float* __restrict__ mp_ss, const float* __restrict__ suppmax,
                              int n){
  int i = blockIdx.x*blockDim.x + threadIdx.x;
  if (i >= n) return;
  bool supp    = suppmax[i] > 0.f;
  bool new_max = (ss[i] == mp_ss[i]) && !supp;
  mask[i] = ((mask[i] > 0.f) || new_max) ? 1.f : 0.f;
}

__global__ void k_final_mask(const float* __restrict__ mask, const float* __restrict__ heat,
                             float* __restrict__ masked){
  int i = blockIdx.x*blockDim.x + threadIdx.x;
  if (i >= 480*640) return;
  int y = i / 640, x = i - y*640;
  float snms = (mask[i] > 0.f) ? heat[i] : 0.f;
  bool valid = (snms > SCORE_THR) && (y >= 4) && (y < 476) && (x >= 4) && (x < 636);
  masked[i] = valid ? snms : 0.f;
}

// per-pixel L2 normalization of 256-channel descriptor map (NCHW f32)
__global__ void k_norm_ad(const float* __restrict__ ad, float* __restrict__ out){
  __shared__ float red[256];
  int p = blockIdx.x;
  int c = threadIdx.x;
  float v = ad[(size_t)c*4800 + p];
  red[c] = v*v;
  __syncthreads();
  for (int s = 128; s > 0; s >>= 1){ if (c < s) red[c] += red[c+s]; __syncthreads(); }
  float nrm = fmaxf(sqrtf(red[0]), 1e-12f);
  out[(size_t)c*4800 + p] = v / nrm;
}

// iterative arg-max top-k (k=1000), ties -> lowest index (matches lax.top_k)
__global__ void k_topk(float* __restrict__ data, float* __restrict__ vals,
                       int* __restrict__ idxs, int n, int kk){
  __shared__ float sv[1024];
  __shared__ int   si[1024];
  int t = threadIdx.x;
  for (int it = 0; it < kk; ++it){
    float bv = -2.f; int bi = 0x7fffffff;
    for (int i = t; i < n; i += 1024){
      float v = data[i];
      if (v > bv || (v == bv && i < bi)){ bv = v; bi = i; }
    }
    sv[t] = bv; si[t] = bi;
    __syncthreads();
    for (int s = 512; s > 0; s >>= 1){
      if (t < s){
        float ov = sv[t+s]; int oi = si[t+s];
        if (ov > sv[t] || (ov == sv[t] && oi < si[t])){ sv[t] = ov; si[t] = oi; }
      }
      __syncthreads();
    }
    if (t == 0){ vals[it] = sv[0]; idxs[it] = si[0]; data[si[0]] = -1.f; }
    __syncthreads();
  }
}

__global__ void k_write_kpts(const float* __restrict__ vals, const int* __restrict__ idxs,
                             float* __restrict__ o_kpts, float* __restrict__ o_scores){
  int i = blockIdx.x*blockDim.x + threadIdx.x;
  if (i >= 1000) return;
  float v = vals[i];
  int id = idxs[i];
  bool good = v > SCORE_THR;
  o_scores[i]    = good ? v : 0.f;
  o_kpts[2*i+0]  = good ? (float)(id % 640) : 0.f;
  o_kpts[2*i+1]  = good ? (float)(id / 640) : 0.f;
}

__device__ __forceinline__ int clampi(int v, int lo, int hi){
  return v < lo ? lo : (v > hi ? hi : v);
}
__device__ __forceinline__ float padread(const float* __restrict__ ad, int c, int yy, int xx){
  if (yy < 1 || yy > 60 || xx < 1 || xx > 80) return 0.f;    // zero pad border
  return ad[((size_t)c*60 + (yy-1))*80 + (xx-1)];
}

// bilinear sample + per-keypoint L2 norm. block = 256 channels, grid = 1000 keypoints
__global__ void k_sample_desc(const float* __restrict__ kpts, const float* __restrict__ ad,
                              float* __restrict__ desc){
  __shared__ float red[256];
  int i = blockIdx.x;
  int c = threadIdx.x;
  float kx = kpts[2*i+0], ky = kpts[2*i+1];
  const int w = 80, h = 60;
  float gx = (kx - 3.5f) / 635.5f * 2.f - 1.f;   // denom = w*8 - 4 - 0.5
  float gy = (ky - 3.5f) / 475.5f * 2.f - 1.f;   // denom = h*8 - 4 - 0.5
  float fx = (gx + 1.f) * 0.5f * (float)(w - 1);
  float fy = (gy + 1.f) * 0.5f * (float)(h - 1);
  int x0 = (int)floorf(fx), y0 = (int)floorf(fy);
  int x1 = x0 + 1, y1 = y0 + 1;
  float wa = ((float)x1 - fx) * ((float)y1 - fy);
  float wb = ((float)x1 - fx) * (fy - (float)y0);
  float wc = (fx - (float)x0) * ((float)y1 - fy);
  float wd = (fx - (float)x0) * (fy - (float)y0);
  int x0c = clampi(x0+1, 0, 81), x1c = clampi(x1+1, 0, 81);
  int y0c = clampi(y0+1, 0, 61), y1c = clampi(y1+1, 0, 61);
  float Ia = padread(ad, c, y0c, x0c);
  float Ib = padread(ad, c, y1c, x0c);
  float Ic = padread(ad, c, y0c, x1c);
  float Id = padread(ad, c, y1c, x1c);
  float v = Ia*wa + Ib*wb + Ic*wc + Id*wd;
  red[c] = v*v;
  __syncthreads();
  for (int s = 128; s > 0; s >>= 1){ if (c < s) red[c] += red[c+s]; __syncthreads(); }
  float nrm = fmaxf(sqrtf(red[0]), 1e-12f);
  desc[(size_t)c*1000 + i] = v / nrm;
}

// ---------------------------------------------------------------- host
extern "C" void kernel_launch(void* const* d_in, const int* in_sizes, int n_in,
                              void* d_out, int out_size, void* d_ws, size_t ws_size,
                              hipStream_t stream){
  (void)in_sizes; (void)n_in; (void)out_size; (void)ws_size;

  const float* image = (const float*)d_in[0];
  const float* w1a = (const float*)d_in[1];  const float* b1a = (const float*)d_in[2];
  const float* w1b = (const float*)d_in[3];  const float* b1b = (const float*)d_in[4];
  const float* w2a = (const float*)d_in[5];  const float* b2a = (const float*)d_in[6];
  const float* w2b = (const float*)d_in[7];  const float* b2b = (const float*)d_in[8];
  const float* w3a = (const float*)d_in[9];  const float* b3a = (const float*)d_in[10];
  const float* w3b = (const float*)d_in[11]; const float* b3b = (const float*)d_in[12];
  const float* w4a = (const float*)d_in[13]; const float* b4a = (const float*)d_in[14];
  const float* w4b = (const float*)d_in[15]; const float* b4b = (const float*)d_in[16];
  const float* wPa = (const float*)d_in[17]; const float* bPa = (const float*)d_in[18];
  const float* wPb = (const float*)d_in[19]; const float* bPb = (const float*)d_in[20];
  const float* wDa = (const float*)d_in[21]; const float* bDa = (const float*)d_in[22];
  const float* wDb = (const float*)d_in[23]; const float* bDb = (const float*)d_in[24];

  char* ws = (char*)d_ws;
  _Float16* bufA   = (_Float16*)(ws + 0);          // 64*480*640 f16 = 39321600 B (NHWC)
  _Float16* bufB   = (_Float16*)(ws + 39321600);   // same
  float*    s0     = (float*)(ws + 78643200);      // 65*4800 f32 (NCHW)
  float*    adraw  = (float*)(ws + 79891200);      // 256*4800 f32 (NCHW)
  float*    heat   = (float*)(ws + 84806400);      // 480*640 f32
  float*    mask   = (float*)(ws + 86035200);
  float*    t0     = (float*)(ws + 87264000);
  float*    t1     = (float*)(ws + 88492800);
  float*    t2     = (float*)(ws + 89721600);
  float*    t3     = (float*)(ws + 90950400);
  float*    masked = (float*)(ws + 92179200);
  float*    vals   = (float*)(ws + 93408000);      // 1000 f32
  int*      idxs   = (int*)  (ws + 93412096);      // 1000 i32
  _Float16* packW  = (_Float16*)(ws + 93417472);   // 1304576 halves

  // packed-weight sub-offsets (halves)
  const size_t P1a=0,      P1b=2048,   P2a=38912,  P2b=75776;
  const size_t P3a=112640, P3b=186368, P4a=333824, P4b=481280;
  const size_t PPa=628736, PPb=923648, PDa=944128, PDb=1239040;

  float* out      = (float*)d_out;
  float* o_kpts   = out;            // (1,1000,2)
  float* o_scores = out + 2000;     // (1,1000)
  float* o_desc   = out + 3000;     // (1,256,1000)
  float* o_adesc  = out + 259000;   // (1,256,60,80)

  auto conv = [&](const _Float16* in, const float* w, const float* bias,
                  _Float16* out16, float* out32, size_t packOff,
                  int Cin, int Cout, int H, int W, int KK, int relu){
    int K = Cin*KK, ksteps = (K+31)/32, mtiles = (Cout+15)/16;
    int total = mtiles*ksteps*512;
    int fast = (Cin % 32 == 0) ? 1 : 0;
    k_pack_w<<<CDIV(total,256), 256, 0, stream>>>(w, packW + packOff, Cout, Cin, KK, ksteps, total);
    dim3 g((H*W)/64, mtiles);
    k_conv_wmma<<<g, 32, 0, stream>>>(in, packW + packOff, bias, out32, out16,
                                      Cin, Cout, H, W, KK, (KK==9)?1:0, ksteps, relu, fast);
  };

  // backbone (activations NHWC f16; image C=1 so layout is identical)
  k_f32_to_f16<<<CDIV(307200,256), 256, 0, stream>>>(image, bufB, 307200);
  conv(bufB, w1a, b1a, bufA, nullptr, P1a,   1,  64, 480, 640, 9, 1);
  conv(bufA, w1b, b1b, bufB, nullptr, P1b,  64,  64, 480, 640, 9, 1);
  k_pool2<<<CDIV(64*240*320,256), 256, 0, stream>>>(bufB, bufA, 64, 240, 320);
  conv(bufA, w2a, b2a, bufB, nullptr, P2a,  64,  64, 240, 320, 9, 1);
  conv(bufB, w2b, b2b, bufA, nullptr, P2b,  64,  64, 240, 320, 9, 1);
  k_pool2<<<CDIV(64*120*160,256), 256, 0, stream>>>(bufA, bufB, 64, 120, 160);
  conv(bufB, w3a, b3a, bufA, nullptr, P3a,  64, 128, 120, 160, 9, 1);
  conv(bufA, w3b, b3b, bufB, nullptr, P3b, 128, 128, 120, 160, 9, 1);
  k_pool2<<<CDIV(128*60*80,256), 256, 0, stream>>>(bufB, bufA, 128, 60, 80);
  conv(bufA, w4a, b4a, bufB, nullptr, P4a, 128, 128, 60, 80, 9, 1);
  conv(bufB, w4b, b4b, bufA, nullptr, P4b, 128, 128, 60, 80, 9, 1);  // x = bufA
  // heads
  conv(bufA, wPa, bPa, bufB, nullptr, PPa, 128, 256, 60, 80, 9, 1);
  conv(bufB, wPb, bPb, nullptr, s0,    PPb, 256,  65, 60, 80, 1, 0);
  conv(bufA, wDa, bDa, bufB, nullptr, PDa, 128, 256, 60, 80, 9, 1);
  conv(bufB, wDb, bDb, nullptr, adraw, PDb, 256, 256, 60, 80, 1, 0);

  // scores: softmax + depth-to-space
  k_softmax65<<<CDIV(4800,256), 256, 0, stream>>>(s0, heat);

  // simple_nms (r=4, 2 iterations) via separable 9x9 maxpools
  const int NPIX = 480*640, NB = CDIV(NPIX,256);
  k_mp_row<<<NB,256,0,stream>>>(heat, t3);
  k_mp_col<<<NB,256,0,stream>>>(t3, t0);
  k_eq_mask<<<NB,256,0,stream>>>(heat, t0, mask, NPIX);
  for (int it = 0; it < 2; ++it){
    k_mp_row<<<NB,256,0,stream>>>(mask, t3);
    k_mp_col<<<NB,256,0,stream>>>(t3, t0);                 // t0 = maxpool(mask)
    k_supp_ss<<<NB,256,0,stream>>>(t0, heat, t1, NPIX);    // t1 = ss
    k_mp_row<<<NB,256,0,stream>>>(t1, t3);
    k_mp_col<<<NB,256,0,stream>>>(t3, t2);                 // t2 = maxpool(ss)
    k_mask_update<<<NB,256,0,stream>>>(mask, t1, t2, t0, NPIX);
  }
  k_final_mask<<<NB,256,0,stream>>>(mask, heat, masked);

  // descriptor map normalization straight into d_out
  k_norm_ad<<<4800, 256, 0, stream>>>(adraw, o_adesc);

  // top-1000 keypoints
  k_topk<<<1, 1024, 0, stream>>>(masked, vals, idxs, NPIX, 1000);
  k_write_kpts<<<4, 256, 0, stream>>>(vals, idxs, o_kpts, o_scores);

  // bilinear descriptor sampling + per-keypoint norm
  k_sample_desc<<<1000, 256, 0, stream>>>(o_kpts, o_adesc, o_desc);
}
